// MultiheadGQA_32856499815032
// MI455X (gfx1250) — matrix operand order repaired and checked by reading, not compiled, not run on hardware
//
#include <hip/hip_runtime.h>
#include <hip/hip_bf16.h>

// ---------------------------------------------------------------------------
// GQA attention layer for MI455X (gfx1250, wave32, WMMA 16x16x32 bf16)
//   y = softmax(mask(rope(xWq) rope(xWk)^T / sqrt(d))) (xWv)  @ Wo
// All GEMM-shaped math runs on v_wmma_f32_16x16x32_bf16 (fp32 accumulate).
// K tiles are staged HBM->LDS with global_load_async_to_lds_b128 (ASYNCcnt),
// LDS tiles are double-buffered, WMMA operand loads are software-pipelined.
// ---------------------------------------------------------------------------

typedef __bf16 bf16_t;
typedef __attribute__((ext_vector_type(16))) __bf16 v16bf;
typedef __attribute__((ext_vector_type(8)))  float  v8f;

struct alignas(16) U128 { unsigned int w[4]; };
union Frag {
    v16bf v;          // 8 VGPRs x 2 bf16 = one A or B fragment (16x32 / 32x16)
    U128  u[2];
    bf16_t h[16];
};

#define BATCH    2
#define TSEQ     2048
#define NEMBD    2048
#define NHEADS   16
#define KVHEADS  4
#define HDIM     128
#define KVWIDTH  (KVHEADS * HDIM)   // 512

// ---------------------------------------------------------------------------
// gfx1250 async HBM->LDS copy (per-lane 16B), tracked by ASYNCcnt.
// ---------------------------------------------------------------------------
typedef __attribute__((address_space(3))) const char* lds_cptr_t;

__device__ __forceinline__ unsigned lds_off(const void* p) {
    return (unsigned)(unsigned long long)(lds_cptr_t)p;   // addrspacecast -> 32-bit LDS offset
}

__device__ __forceinline__ void async_ld_b128(unsigned lds_addr, const void* gaddr) {
    asm volatile("global_load_async_to_lds_b128 %0, %1, off"
                 :: "v"(lds_addr), "v"((unsigned long long)(size_t)gaddr)
                 : "memory");
}

__device__ __forceinline__ void wait_async0() {
    asm volatile("s_wait_asynccnt 0x0" ::: "memory");
}

// ---------------------------------------------------------------------------
// Elementwise fp32 -> bf16 cast
// ---------------------------------------------------------------------------
__global__ void cast_f32_bf16_kernel(const float* __restrict__ in,
                                     bf16_t* __restrict__ out, int n) {
    int i = blockIdx.x * blockDim.x + threadIdx.x;
    int stride = gridDim.x * blockDim.x;
    for (; i < n; i += stride) out[i] = (bf16_t)in[i];
}

// ---------------------------------------------------------------------------
// RoPE (pairs (2i, 2i+1) within each head), fp32 in -> bf16 out.
// ---------------------------------------------------------------------------
__global__ void rope_bf16_kernel(const float* __restrict__ in,
                                 bf16_t* __restrict__ out,
                                 int rows, int heads) {
    const int pairsPerRow = heads * (HDIM / 2);
    int total = rows * pairsPerRow;
    int idx = blockIdx.x * blockDim.x + threadIdx.x;
    int stride = gridDim.x * blockDim.x;
    for (; idx < total; idx += stride) {
        int row = idx / pairsPerRow;
        int p   = idx - row * pairsPerRow;
        int h   = p >> 6;          // p / 64
        int i   = p & 63;          // pair index within head
        int t   = row & (TSEQ - 1);
        float inv_freq = __expf(-(float)(2 * i) * 0.07195578428466385f); // ln(1e4)/128
        float ang = (float)t * inv_freq;
        float sn, cs;
        __sincosf(ang, &sn, &cs);
        int width = heads * HDIM;
        size_t base = (size_t)row * width + h * HDIM + 2 * i;
        float e0 = in[base], e1 = in[base + 1];
        out[base]     = (bf16_t)(e0 * cs - e1 * sn);
        out[base + 1] = (bf16_t)(e0 * sn + e1 * cs);
    }
}

// ---------------------------------------------------------------------------
// GEMM: C[M,N] (fp32) = A[M,K] (bf16) * B[K,N] (bf16), both row major.
// Block = 256 threads = 8 waves; block tile 128x64, wave tile 16x64.
// B tile staged TRANSPOSED ([n][k]) in double-buffered LDS; the next tile's
// global load is issued before the current tile's WMMAs (latency hiding),
// and all 4 B fragments are preloaded so the 4 WMMAs issue back-to-back.
// ---------------------------------------------------------------------------
__global__ __launch_bounds__(256)
void gemm_bf16_wmma_kernel(const bf16_t* __restrict__ A,
                           const bf16_t* __restrict__ B,
                           float* __restrict__ C,
                           int M, int N, int K) {
    __shared__ bf16_t Bt[2][64 * 32];   // [buf][n][k], 2 x 4 KB

    const int lane = threadIdx.x & 31;
    const int wave = threadIdx.x >> 5;
    const int n16  = lane & 15;
    const int half = lane >> 4;

    const int nBlocks = N >> 6;
    const int bm = blockIdx.x / nBlocks;
    const int bn = blockIdx.x - bm * nBlocks;
    const int m0 = bm * 128 + wave * 16;
    const int n0 = bn * 64;

    const int tid = threadIdx.x;
    const int kr  = tid >> 3;   // 0..31 : k row of B tile
    const int cg  = tid & 7;    // 8 bf16 columns per thread
    const bf16_t* bsrc = B + (size_t)kr * N + n0 + cg * 8;

    // ---- prologue: stage tile 0 into buffer 0 ----
    {
        Frag tmp;
        tmp.u[0] = *(const U128*)(bsrc);
        #pragma unroll
        for (int e = 0; e < 8; ++e)
            Bt[0][(cg * 8 + e) * 32 + kr] = tmp.h[e];
    }
    __syncthreads();

    v8f acc[4];
    #pragma unroll
    for (int c = 0; c < 4; ++c)
        #pragma unroll
        for (int r = 0; r < 8; ++r) acc[c][r] = 0.0f;

    for (int k0 = 0; k0 < K; k0 += 32) {
        const int cur = (k0 >> 5) & 1;
        const bool haveNext = (k0 + 32 < K);

        // issue next B tile's global load early (overlaps with WMMAs below)
        Frag nxt;
        if (haveNext) {
            nxt.u[0] = *(const U128*)(bsrc + (size_t)(k0 + 32) * N);
            if (k0 + 64 < K)
                __builtin_prefetch(bsrc + (size_t)(k0 + 64) * N, 0, 1);
        }

        // A fragment: two contiguous 16B global loads per lane
        Frag af;
        const bf16_t* arow = A + (size_t)(m0 + n16) * K + k0;
        af.u[0] = *(const U128*)(arow + half * 8);
        af.u[1] = *(const U128*)(arow + 16 + half * 8);

        // preload all 4 B fragments, then 4 back-to-back WMMAs
        Frag bfr[4];
        #pragma unroll
        for (int c = 0; c < 4; ++c) {
            const bf16_t* bcol = &Bt[cur][(c * 16 + n16) * 32 + half * 8];
            bfr[c].u[0] = *(const U128*)(bcol);
            bfr[c].u[1] = *(const U128*)(bcol + 16);
        }
        #pragma unroll
        for (int c = 0; c < 4; ++c)
            acc[c] = __builtin_amdgcn_wmma_f32_16x16x32_bf16(
                false, af.v, false, bfr[c].v, (short)0, acc[c], false, false);

        // stage next tile into the other buffer
        if (haveNext) {
            #pragma unroll
            for (int e = 0; e < 8; ++e)
                Bt[cur ^ 1][(cg * 8 + e) * 32 + kr] = nxt.h[e];
        }
        __syncthreads();
    }

    // ---- store: C layout is VGPR r -> row r + half*8, lane n16 -> col ----
    #pragma unroll
    for (int c = 0; c < 4; ++c)
        #pragma unroll
        for (int r = 0; r < 8; ++r)
            C[(size_t)(m0 + r + half * 8) * N + n0 + c * 16 + n16] = acc[c][r];
}

// ---------------------------------------------------------------------------
// Flash-attention (causal, GQA). One block = (b, head, 128 query rows),
// 8 waves x 16-row query tiles. Keys processed 32 at a time.
//  - Q fragments (16x128 = 4 k-chunks) held in registers for whole loop.
//  - K tile [32][128] staged via global_load_async_to_lds_b128 (ASYNCcnt).
//  - V tile transposed [128][32] via VGPR staging (overlapped with compute).
//  - K/V LDS tiles double-buffered; next tile staged while computing current.
//  - P converted C-layout -> A-layout via per-wave LDS scratch.
// ---------------------------------------------------------------------------
__global__ __launch_bounds__(256)
void attn_fa_kernel(const bf16_t* __restrict__ qb,
                    const bf16_t* __restrict__ kb,
                    const bf16_t* __restrict__ vb,
                    bf16_t* __restrict__ yb) {
    __shared__ bf16_t Kt[2][32 * 128];    // 2 x 8 KB   [s][d]
    __shared__ bf16_t Vt[2][128 * 32];    // 2 x 8 KB   [d][s]
    __shared__ bf16_t Pls[8][16 * 32];    // 8 KB       per-wave P scratch [m][s]

    const int lane = threadIdx.x & 31;
    const int wave = threadIdx.x >> 5;
    const int n16  = lane & 15;
    const int half = lane >> 4;

    const int qblk = blockIdx.x & 15;
    const int h    = (blockIdx.x >> 4) & 15;
    const int b    = blockIdx.x >> 8;
    const int kvh  = h >> 2;                      // REPEATS = 4
    const int q0   = qblk * 128 + wave * 16;
    const int tmaxrow = q0 + 15;

    const int tid = threadIdx.x;
    const int kr  = tid >> 3;       // key row 0..31
    const int cg  = tid & 7;        // 16 bf16 along d per thread
    const bf16_t* kbase = kb + (size_t)(b * TSEQ + kr) * KVWIDTH + kvh * HDIM + cg * 16;
    const bf16_t* vbase = vb + (size_t)(b * TSEQ + kr) * KVWIDTH + kvh * HDIM + cg * 16;

    // ---- Q fragments: 4 chunks of K=32 along d ----
    Frag qf[4];
    {
        const bf16_t* qrow = qb + (size_t)(b * TSEQ + q0 + n16) * NEMBD + h * HDIM;
        #pragma unroll
        for (int kc = 0; kc < 4; ++kc) {
            qf[kc].u[0] = *(const U128*)(qrow + kc * 32 + half * 8);
            qf[kc].u[1] = *(const U128*)(qrow + kc * 32 + 16 + half * 8);
        }
    }

    float m_run[8], l_run[8];
    v8f O[8];
    #pragma unroll
    for (int r = 0; r < 8; ++r) { m_run[r] = -1e30f; l_run[r] = 0.0f; }
    #pragma unroll
    for (int c = 0; c < 8; ++c)
        #pragma unroll
        for (int r = 0; r < 8; ++r) O[c][r] = 0.0f;

    const float SCALE = 0.08838834764831845f;     // 1/sqrt(128)
    const int sEnd = qblk * 128 + 128;

    // ---- prologue: stage key block 0 into buffer 0 ----
    {
        async_ld_b128(lds_off(&Kt[0][kr * 128 + cg * 16]),     kbase);
        async_ld_b128(lds_off(&Kt[0][kr * 128 + cg * 16 + 8]), kbase + 8);
        Frag vtmp;
        vtmp.u[0] = *(const U128*)(vbase);
        vtmp.u[1] = *(const U128*)(vbase + 8);
        #pragma unroll
        for (int e = 0; e < 16; ++e)
            Vt[0][(cg * 16 + e) * 32 + kr] = vtmp.h[e];
    }
    wait_async0();
    __syncthreads();

    for (int s0 = 0; s0 < sEnd; s0 += 32) {
        const int cur = (s0 >> 5) & 1;
        const bool haveNext = (s0 + 32 < sEnd);

        // ---- stage NEXT key block: async K -> LDS, V -> VGPRs ----
        Frag vstage;
        if (haveNext) {
            const size_t soff = (size_t)(s0 + 32) * KVWIDTH;
            async_ld_b128(lds_off(&Kt[cur ^ 1][kr * 128 + cg * 16]),     kbase + soff);
            async_ld_b128(lds_off(&Kt[cur ^ 1][kr * 128 + cg * 16 + 8]), kbase + soff + 8);
            vstage.u[0] = *(const U128*)(vbase + soff);
            vstage.u[1] = *(const U128*)(vbase + soff + 8);
        }

        if (s0 <= tmaxrow) {   // wave-uniform: EXEC stays all-ones for WMMA
            // ---- scores: two 16x16 tiles, K=128 via 4 WMMA steps each ----
            v8f S0, S1;
            #pragma unroll
            for (int r = 0; r < 8; ++r) { S0[r] = 0.0f; S1[r] = 0.0f; }
            #pragma unroll
            for (int kc = 0; kc < 4; ++kc) {
                Frag kf0, kf1;
                const bf16_t* kp0 = &Kt[cur][n16 * 128 + kc * 32 + half * 8];
                const bf16_t* kp1 = kp0 + 16 * 128;
                kf0.u[0] = *(const U128*)(kp0);
                kf0.u[1] = *(const U128*)(kp0 + 16);
                kf1.u[0] = *(const U128*)(kp1);
                kf1.u[1] = *(const U128*)(kp1 + 16);
                S0 = __builtin_amdgcn_wmma_f32_16x16x32_bf16(
                    false, qf[kc].v, false, kf0.v, (short)0, S0, false, false);
                S1 = __builtin_amdgcn_wmma_f32_16x16x32_bf16(
                    false, qf[kc].v, false, kf1.v, (short)0, S1, false, false);
            }

            // ---- scale + causal mask + row max ----
            float rmax[8];
            #pragma unroll
            for (int r = 0; r < 8; ++r) {
                const int trow = q0 + r + half * 8;
                float a0 = S0[r] * SCALE;
                float a1 = S1[r] * SCALE;
                if (s0 + n16 > trow)      a0 = -1e30f;
                if (s0 + 16 + n16 > trow) a1 = -1e30f;
                S0[r] = a0; S1[r] = a1;
                rmax[r] = fmaxf(a0, a1);
            }
            #pragma unroll
            for (int r = 0; r < 8; ++r)
                #pragma unroll
                for (int mk = 1; mk < 16; mk <<= 1)
                    rmax[r] = fmaxf(rmax[r], __shfl_xor(rmax[r], mk, 32));

            // ---- online softmax update ----
            float alpha[8], rsum[8];
            #pragma unroll
            for (int r = 0; r < 8; ++r) {
                float mn = fmaxf(m_run[r], rmax[r]);
                alpha[r] = __expf(m_run[r] - mn);
                m_run[r] = mn;
                float p0 = __expf(S0[r] - mn);
                float p1 = __expf(S1[r] - mn);
                S0[r] = p0; S1[r] = p1;
                rsum[r] = p0 + p1;
            }
            #pragma unroll
            for (int r = 0; r < 8; ++r)
                #pragma unroll
                for (int mk = 1; mk < 16; mk <<= 1)
                    rsum[r] += __shfl_xor(rsum[r], mk, 32);
            #pragma unroll
            for (int r = 0; r < 8; ++r) l_run[r] = l_run[r] * alpha[r] + rsum[r];
            #pragma unroll
            for (int c = 0; c < 8; ++c)
                #pragma unroll
                for (int r = 0; r < 8; ++r) O[c][r] *= alpha[r];

            // ---- P: C-layout -> LDS -> A-layout fragment ----
            bf16_t* pw = &Pls[wave][0];
            #pragma unroll
            for (int r = 0; r < 8; ++r) {
                pw[(r + half * 8) * 32 + n16]      = (bf16_t)S0[r];
                pw[(r + half * 8) * 32 + 16 + n16] = (bf16_t)S1[r];
            }
            Frag pf;   // same-wave LDS RAW; DScnt wait inserted by compiler
            pf.u[0] = *(const U128*)(&pw[n16 * 32 + half * 8]);
            pf.u[1] = *(const U128*)(&pw[n16 * 32 + 16 + half * 8]);

            // ---- O += P (16x32) * V (32x16 per d-chunk); pipelined V frags ----
            Frag vf[2];
            {
                const bf16_t* vp = &Vt[cur][n16 * 32 + half * 8];
                vf[0].u[0] = *(const U128*)(vp);
                vf[0].u[1] = *(const U128*)(vp + 16);
            }
            #pragma unroll
            for (int c = 0; c < 8; ++c) {
                if (c < 7) {
                    const bf16_t* vp = &Vt[cur][((c + 1) * 16 + n16) * 32 + half * 8];
                    vf[(c + 1) & 1].u[0] = *(const U128*)(vp);
                    vf[(c + 1) & 1].u[1] = *(const U128*)(vp + 16);
                }
                O[c] = __builtin_amdgcn_wmma_f32_16x16x32_bf16(
                    false, pf.v, false, vf[c & 1].v, (short)0, O[c], false, false);
            }
        }

        // ---- finish staging next block, then block-wide barrier ----
        if (haveNext) {
            #pragma unroll
            for (int e = 0; e < 16; ++e)
                Vt[cur ^ 1][(cg * 16 + e) * 32 + kr] = vstage.h[e];
        }
        wait_async0();
        __syncthreads();
    }

    // ---- normalize and store y (bf16) for the Wo projection ----
    float invl[8];
    #pragma unroll
    for (int r = 0; r < 8; ++r) invl[r] = 1.0f / l_run[r];
    #pragma unroll
    for (int c = 0; c < 8; ++c)
        #pragma unroll
        for (int r = 0; r < 8; ++r) {
            size_t idx = (size_t)(b * TSEQ + q0 + r + half * 8) * NEMBD
                       + h * HDIM + c * 16 + n16;
            yb[idx] = (bf16_t)(O[c][r] * invl[r]);
        }
}

// ---------------------------------------------------------------------------
// Host-side orchestration
// ---------------------------------------------------------------------------
extern "C" void kernel_launch(void* const* d_in, const int* in_sizes, int n_in,
                              void* d_out, int out_size, void* d_ws, size_t ws_size,
                              hipStream_t stream) {
    const float* x  = (const float*)d_in[0];
    const float* wq = (const float*)d_in[1];
    const float* wk = (const float*)d_in[2];
    const float* wv = (const float*)d_in[3];
    const float* wo = (const float*)d_in[4];
    float* out = (float*)d_out;

    const int BT = BATCH * TSEQ;                      // 4096

    char* ws = (char*)d_ws;
    size_t off = 0;
    auto alloc = [&](size_t bytes) -> void* {
        void* p = ws + off;
        off = (off + bytes + 255) & ~(size_t)255;
        return p;
    };

    bf16_t* xb   = (bf16_t*)alloc((size_t)BT * NEMBD * 2);
    bf16_t* wqb  = (bf16_t*)alloc((size_t)NEMBD * NEMBD * 2);
    bf16_t* wkb  = (bf16_t*)alloc((size_t)NEMBD * KVWIDTH * 2);
    bf16_t* wvb  = (bf16_t*)alloc((size_t)NEMBD * KVWIDTH * 2);
    bf16_t* wob  = (bf16_t*)alloc((size_t)NEMBD * NEMBD * 2);
    float*  qf32 = (float*)alloc((size_t)BT * NEMBD * 4);
    float*  kf32 = (float*)alloc((size_t)BT * KVWIDTH * 4);
    float*  vf32 = (float*)alloc((size_t)BT * KVWIDTH * 4);
    bf16_t* qbf  = (bf16_t*)alloc((size_t)BT * NEMBD * 2);
    bf16_t* kbf  = (bf16_t*)alloc((size_t)BT * KVWIDTH * 2);
    bf16_t* vbf  = (bf16_t*)alloc((size_t)BT * KVWIDTH * 2);
    bf16_t* ybf  = (bf16_t*)alloc((size_t)BT * NEMBD * 2);

    const dim3 blk(256);

    // 1) casts to bf16
    cast_f32_bf16_kernel<<<2048, blk, 0, stream>>>(x,  xb,  BT * NEMBD);
    cast_f32_bf16_kernel<<<2048, blk, 0, stream>>>(wq, wqb, NEMBD * NEMBD);
    cast_f32_bf16_kernel<<<1024, blk, 0, stream>>>(wk, wkb, NEMBD * KVWIDTH);
    cast_f32_bf16_kernel<<<1024, blk, 0, stream>>>(wv, wvb, NEMBD * KVWIDTH);
    cast_f32_bf16_kernel<<<2048, blk, 0, stream>>>(wo, wob, NEMBD * NEMBD);

    // 2) QKV projections (WMMA GEMM)
    gemm_bf16_wmma_kernel<<<(BT / 128) * (NEMBD   / 64), blk, 0, stream>>>(xb, wqb, qf32, BT, NEMBD,   NEMBD);
    gemm_bf16_wmma_kernel<<<(BT / 128) * (KVWIDTH / 64), blk, 0, stream>>>(xb, wkb, kf32, BT, KVWIDTH, NEMBD);
    gemm_bf16_wmma_kernel<<<(BT / 128) * (KVWIDTH / 64), blk, 0, stream>>>(xb, wvb, vf32, BT, KVWIDTH, NEMBD);

    // 3) RoPE for q,k; plain cast for v
    rope_bf16_kernel<<<4096, blk, 0, stream>>>(qf32, qbf, BT, NHEADS);
    rope_bf16_kernel<<<1024, blk, 0, stream>>>(kf32, kbf, BT, KVHEADS);
    cast_f32_bf16_kernel<<<1024, blk, 0, stream>>>(vf32, vbf, BT * KVWIDTH);

    // 4) flash attention: B * NHEADS * (T/128) blocks = 2*16*16 = 512
    attn_fa_kernel<<<BATCH * NHEADS * (TSEQ / 128), blk, 0, stream>>>(qbf, kbf, vbf, ybf);

    // 5) output projection straight to d_out (fp32)
    gemm_bf16_wmma_kernel<<<(BT / 128) * (NEMBD / 64), blk, 0, stream>>>(ybf, wob, out, BT, NEMBD, NEMBD);
}
